// PointConv_42073499631902
// MI455X (gfx1250) — matrix operand order repaired and checked by reading, not compiled, not run on hardware
//
#include <hip/hip_runtime.h>
#include <cstdint>
#include <cstddef>

// Problem constants
#define B_    8
#define N_    8192
#define S_    2048
#define K_    32
#define D_    128
#define C0_   131          // D + 3 input channels
#define C0P_  160          // padded to multiple of 32 for WMMA K-chunks
#define M1_   128
#define M2_   128
#define M3_   256
#define HID_  8
#define LEAKY_ 0.1f
#define FLATC_ (M3_ * HID_)   // 2048

#define W1N (M1_ * C0P_)
#define W2N (M2_ * M1_)
#define W3N (M3_ * M2_)
#define WLN (M3_ * FLATC_)
#define PTN (B_ * N_ * D_)    // transposed bf16 points

typedef __attribute__((ext_vector_type(16))) __bf16       v16bf;
typedef __attribute__((ext_vector_type(8)))  float        v8f;
typedef __attribute__((ext_vector_type(4)))  unsigned int v4u;

static __device__ __forceinline__ __bf16 f2bf(float f) {
  unsigned int u = __builtin_bit_cast(unsigned int, f);
  unsigned int r = u + 0x7FFFu + ((u >> 16) & 1u);   // RNE
  unsigned short h = (unsigned short)(r >> 16);
  return __builtin_bit_cast(__bf16, h);
}
static __device__ __forceinline__ float bf2f(__bf16 x) {
  unsigned short h = __builtin_bit_cast(unsigned short, x);
  unsigned int u = ((unsigned int)h) << 16;
  return __builtin_bit_cast(float, u);
}
static __device__ __forceinline__ float lrelu(float v) { return v > 0.f ? v : LEAKY_ * v; }

// ---- WMMA fragment helpers (A/B: same contiguous K order; C layout per CDNA5 ISA §7.12.2) ----

// A (16x32 bf16): lanes 0-15 row=lane, K 0-15; lanes 16-31 row=lane-16, K 16-31
static __device__ __forceinline__ v16bf load_a_frag(const __bf16* tile, int stride, int lane) {
  int row = lane & 15;
  int k0  = (lane & 16) ? 16 : 0;
  const __bf16* p = tile + row * stride + k0;
  union { v16bf v; v4u q[2]; } r;
  r.q[0] = *(const v4u*)(p);        // K k0..k0+7
  r.q[1] = *(const v4u*)(p + 8);    // K k0+8..k0+15
  return r.v;
}

// B (32x16 bf16), weights stored row-major [out][in] => per-lane K is contiguous
static __device__ __forceinline__ v16bf load_b_frag(const __bf16* w, int n0, int stride, int kc, int lane) {
  int n = n0 + (lane & 15);
  int k = kc + ((lane & 16) ? 16 : 0);
  return *(const v16bf*)(w + (size_t)n * stride + k);
}

static __device__ __forceinline__ v8f wmma_bf16(v16bf a, v16bf b, v8f c) {
  // (neg_a, A, neg_b, B, c_mod, C, reuse_a, reuse_b)
  return __builtin_amdgcn_wmma_f32_16x16x32_bf16(false, a, false, b, (short)0, c, false, false);
}

// C (16x16 f32): VGPR r -> M = r + (lane[4]?8:0), N = lane&15
static __device__ __forceinline__ void store_c_lrelu_lds(v8f c, __bf16* dst, int stride, int lane) {
  int m0 = (lane & 16) ? 8 : 0;
  int n  = lane & 15;
#pragma unroll
  for (int r = 0; r < 8; ++r)
    dst[(m0 + r) * stride + n] = f2bf(lrelu(c[r]));
}

// One MLP layer for a 32-row activation tile: out = lrelu(in(32 x cin) * wgt^T(cin x cout))
static __device__ __forceinline__ void mlp_layer(const __bf16* in, int cin,
                                                 const __bf16* wgt, int cout,
                                                 __bf16* out, int wave, int lane) {
  const int ntiles = cout >> 4;
  for (int nt = wave; nt < ntiles; nt += 8) {
    v8f acc0 = {}, acc1 = {};
    for (int kc = 0; kc < cin; kc += 32) {
      v16bf bf = load_b_frag(wgt, nt * 16, cin, kc, lane);
      v16bf a0 = load_a_frag(in + kc, cin, lane);             // rows 0..15
      v16bf a1 = load_a_frag(in + 16 * cin + kc, cin, lane);  // rows 16..31
      acc0 = wmma_bf16(a0, bf, acc0);
      acc1 = wmma_bf16(a1, bf, acc1);
    }
    store_c_lrelu_lds(acc0, out + nt * 16, cout, lane);
    store_c_lrelu_lds(acc1, out + 16 * cout + nt * 16, cout, lane);
  }
}

// ---------------- Kernel 1: new_xyz = xyz[:, :, :S] ----------------
__global__ __launch_bounds__(256) void pc_newxyz_kernel(const float* __restrict__ xyz,
                                                        float* __restrict__ out) {
  int i = blockIdx.x * blockDim.x + threadIdx.x;
  if (i >= B_ * 3 * S_) return;
  int b = i / (3 * S_);
  int r = i - b * 3 * S_;
  int c = r / S_;
  int s = r - c * S_;
  out[i] = xyz[(size_t)b * 3 * N_ + (size_t)c * N_ + s];
}

// ---------------- Kernel 2: weights -> bf16 (w1 columns permuted: [points, dir]) --------
__global__ __launch_bounds__(256) void pc_prep_kernel(const float* __restrict__ w1,
                                                      const float* __restrict__ w2,
                                                      const float* __restrict__ w3,
                                                      const float* __restrict__ wlin,
                                                      __bf16* __restrict__ w1b, __bf16* __restrict__ w2b,
                                                      __bf16* __restrict__ w3b, __bf16* __restrict__ wlinb) {
  int i = blockIdx.x * blockDim.x + threadIdx.x;
  if (i < W1N) {
    int o = i / C0P_, c = i - o * C0P_;
    // feats channel order in LDS: [grouped 0..127, dir 128..130, pad]
    float v;
    if (c < D_)            v = w1[o * C0_ + 3 + c];      // grouped points part
    else if (c < C0_)      v = w1[o * C0_ + (c - D_)];   // direction part
    else                   v = 0.f;                      // pad
    w1b[i] = f2bf(v);
  } else if (i < W1N + W2N) {
    int j = i - W1N; w2b[j] = f2bf(w2[j]);
  } else if (i < W1N + W2N + W3N) {
    int j = i - W1N - W2N; w3b[j] = f2bf(w3[j]);
  } else if (i < W1N + W2N + W3N + WLN) {
    int j = i - W1N - W2N - W3N; wlinb[j] = f2bf(wlin[j]);
  }
}

// ---------------- Kernel 2b: points (B,D,N) f32 -> points_t (B,N,D) bf16 ----------------
__global__ __launch_bounds__(256) void pc_transpose_kernel(const float* __restrict__ points,
                                                           __bf16* __restrict__ points_t) {
  int i = blockIdx.x * blockDim.x + threadIdx.x;
  if (i >= PTN) return;
  int b = i / (N_ * D_);
  int r = i - b * N_ * D_;
  int n = r >> 7;          // / D_
  int d = r & (D_ - 1);
  points_t[i] = f2bf(points[((size_t)b * D_ + d) * N_ + n]);
}

// ---------------- Kernel 3: KNN with LDS candidate tiling ----------------
#define KT_ 512
__global__ __launch_bounds__(256) void pc_knn_kernel(const float* __restrict__ xyz,
                                                     int* __restrict__ knn) {
  __shared__ float sx[KT_], sy[KT_], sz[KT_];
  const int tid = threadIdx.x;
  const int q = blockIdx.x * 256 + tid;   // S per batch is a multiple of 256 -> block-uniform b
  const int b = q / S_;
  const int s = q - b * S_;
  const float* xb = xyz + (size_t)b * 3 * N_;
  const float qx = xb[s], qy = xb[N_ + s], qz = xb[2 * N_ + s];
  float bd[K_];
  int   bi[K_];
#pragma unroll
  for (int j = 0; j < K_; ++j) { bd[j] = 3.0e38f; bi[j] = 0; }
  float worst = 3.0e38f;
  int wslot = 0;
  for (int t0 = 0; t0 < N_; t0 += KT_) {
    __syncthreads();
    for (int j = tid; j < KT_; j += 256) {
      sx[j] = xb[t0 + j];
      sy[j] = xb[N_ + t0 + j];
      sz[j] = xb[2 * N_ + t0 + j];
    }
    __syncthreads();
    for (int j = 0; j < KT_; ++j) {
      float dx = sx[j] - qx;
      float dy = sy[j] - qy;
      float dz = sz[j] - qz;
      float d = dx * dx + dy * dy + dz * dz;
      if (d < worst) {
        bd[wslot] = d; bi[wslot] = t0 + j;
        worst = -1.f;
#pragma unroll
        for (int u = 0; u < K_; ++u)
          if (bd[u] > worst) { worst = bd[u]; wslot = u; }
      }
    }
  }
  int* op = knn + (size_t)q * K_;
#pragma unroll
  for (int j = 0; j < K_; ++j) op[j] = bi[j];
}

// ------- Kernel 4: async gather + 3-layer WMMA MLP + weight-net + K-reduction -------
__global__ __launch_bounds__(256) void pc_mlp_kernel(
    const float* __restrict__ xyz, const __bf16* __restrict__ points_t,
    const int* __restrict__ knn,
    const __bf16* __restrict__ w1b, const __bf16* __restrict__ w2b, const __bf16* __restrict__ w3b,
    const float* __restrict__ wn_w1, const float* __restrict__ wn_b1,
    const float* __restrict__ wn_w2, const float* __restrict__ wn_b2,
    const float* __restrict__ wn_w3, const float* __restrict__ wn_b3,
    __bf16* __restrict__ flat)
{
  __shared__ __align__(32) __bf16 actA[K_ * C0P_];  // L1 in (stride 160) / L2 out, L3 in (stride 128)
  __shared__ __align__(32) __bf16 actB[K_ * M3_];   // L1 out, L2 in (stride 128) / L3 out (stride 256)
  __shared__ float dirS[K_ * 4];
  __shared__ float wS[K_ * HID_];

  const int tid  = threadIdx.x;
  const int wave = tid >> 5;
  const int lane = tid & 31;
  const int q = blockIdx.x;            // b*S + s
  const int b = q / S_;
  const int s = q - b * S_;
  const float* xb = xyz + (size_t)b * 3 * N_;
  const int* kq = knn + (size_t)q * K_;

  // --- stage feats: channels [0..127]=grouped (async DMA), [128..130]=dir, [131..159]=0 ---
  // zero the tail region (dir gets overwritten below)
  for (int e = tid; e < K_ * 32; e += 256) {
    int k = e >> 5, c = C0P_ - 32 + (e & 31);
    actA[k * C0P_ + c] = __builtin_bit_cast(__bf16, (unsigned short)0);
  }
  // direction channels + float copy for the weight-net
  const float qx = xb[s], qy = xb[N_ + s], qz = xb[2 * N_ + s];
  if (tid < K_ * 3) {
    int k = tid / 3, c = tid - k * 3;
    int idx = kq[k];
    float v = xb[c * N_ + idx] - (c == 0 ? qx : (c == 1 ? qy : qz));
    dirS[k * 4 + c] = v;
    actA[k * C0P_ + D_ + c] = f2bf(v);
  }
  // async gather: 32 neighbors x 256B each = 512 x B128 transfers, 2 per thread
  {
    const __bf16* ptb = points_t + (size_t)b * N_ * D_;
#pragma unroll
    for (int rr = 0; rr < 2; ++rr) {
      int e = tid + rr * 256;            // 0..511
      int k = e >> 4;                    // neighbor
      int j = e & 15;                    // 16B chunk within 256B row
      int idx = kq[k];
      const __bf16* g = ptb + (size_t)idx * D_ + j * 8;
      unsigned int l = (unsigned int)(uintptr_t)(actA + k * C0P_ + j * 8);
      asm volatile("global_load_async_to_lds_b128 %0, %1, off"
                   :: "v"(l), "v"(g) : "memory");
    }
    asm volatile("s_wait_asynccnt 0" ::: "memory");
  }
  __syncthreads();

  // tiny weight-net on direction (wave 0, one lane per neighbor) — fp32 VALU
  if (tid < K_) {
    float d0 = dirS[tid * 4 + 0], d1 = dirS[tid * 4 + 1], d2 = dirS[tid * 4 + 2];
    float h1[HID_], h2[HID_];
#pragma unroll
    for (int o = 0; o < HID_; ++o) {
      float a = wn_w1[o * 3 + 0] * d0 + wn_w1[o * 3 + 1] * d1 + wn_w1[o * 3 + 2] * d2 + wn_b1[o];
      h1[o] = a > 0.f ? a : 0.f;
    }
#pragma unroll
    for (int o = 0; o < HID_; ++o) {
      float a = wn_b2[o];
#pragma unroll
      for (int i = 0; i < HID_; ++i) a += wn_w2[o * HID_ + i] * h1[i];
      h2[o] = a > 0.f ? a : 0.f;
    }
#pragma unroll
    for (int o = 0; o < HID_; ++o) {
      float a = wn_b3[o];
#pragma unroll
      for (int i = 0; i < HID_; ++i) a += wn_w3[o * HID_ + i] * h2[i];
      wS[tid * HID_ + o] = a > 0.f ? a : 0.f;
    }
  }

  // 3 WMMA layers (EXEC all-ones in every wave at WMMA sites)
  mlp_layer(actA, C0P_, w1b, M1_, actB, wave, lane);
  __syncthreads();
  mlp_layer(actB, M1_, w2b, M2_, actA, wave, lane);
  __syncthreads();
  mlp_layer(actA, M2_, w3b, M3_, actB, wave, lane);
  __syncthreads();

  // agg[o][hw] = sum_k h[k][o] * w[k][hw]; one out-channel per thread
  {
    const int o = tid;   // 0..255
    float acc[HID_];
#pragma unroll
    for (int hw = 0; hw < HID_; ++hw) acc[hw] = 0.f;
    for (int k = 0; k < K_; ++k) {
      float hv = bf2f(actB[k * M3_ + o]);
#pragma unroll
      for (int hw = 0; hw < HID_; ++hw) acc[hw] += hv * wS[k * HID_ + hw];
    }
    __bf16* fp = flat + (size_t)q * FLATC_ + o * HID_;
#pragma unroll
    for (int hw = 0; hw < HID_; ++hw) fp[hw] = f2bf(acc[hw]);
  }
}

// ---------------- Kernel 5: out = lrelu(flat @ wlin^T), WMMA over K=2048 ----------------
__global__ __launch_bounds__(256) void pc_final_kernel(const __bf16* __restrict__ flat,
                                                       const __bf16* __restrict__ wlinb,
                                                       float* __restrict__ out) {
  const int wave = threadIdx.x >> 5;
  const int lane = threadIdx.x & 31;
  const int q0 = blockIdx.x * 16;          // 16 queries per block (same batch: S multiple of 16)
  const int b  = q0 / S_;
  const int s0 = q0 - b * S_;
  const __bf16* abase = flat + (size_t)q0 * FLATC_;
  const int nt0 = wave, nt1 = wave + 8;    // two 16-wide channel tiles per wave
  v8f acc0 = {}, acc1 = {};
  for (int kc = 0; kc < FLATC_; kc += 32) {
    v16bf a  = load_a_frag(abase + kc, FLATC_, lane);
    v16bf b0 = load_b_frag(wlinb, nt0 * 16, FLATC_, kc, lane);
    v16bf b1 = load_b_frag(wlinb, nt1 * 16, FLATC_, kc, lane);
    acc0 = wmma_bf16(a, b0, acc0);
    acc1 = wmma_bf16(a, b1, acc1);
  }
  const int m0 = (lane & 16) ? 8 : 0;
  const int n  = lane & 15;
  float* ob = out + (size_t)b * M3_ * S_;
#pragma unroll
  for (int r = 0; r < 8; ++r) {
    int sq = s0 + m0 + r;
    ob[(size_t)(nt0 * 16 + n) * S_ + sq] = lrelu(acc0[r]);
    ob[(size_t)(nt1 * 16 + n) * S_ + sq] = lrelu(acc1[r]);
  }
}

extern "C" void kernel_launch(void* const* d_in, const int* in_sizes, int n_in,
                              void* d_out, int out_size, void* d_ws, size_t ws_size,
                              hipStream_t stream) {
  (void)in_sizes; (void)n_in; (void)out_size; (void)ws_size;
  const float* xyz    = (const float*)d_in[0];
  const float* points = (const float*)d_in[1];
  const float* w1     = (const float*)d_in[2];
  const float* w2     = (const float*)d_in[3];
  const float* w3     = (const float*)d_in[4];
  const float* wn_w1  = (const float*)d_in[5];
  const float* wn_b1  = (const float*)d_in[6];
  const float* wn_w2  = (const float*)d_in[7];
  const float* wn_b2  = (const float*)d_in[8];
  const float* wn_w3  = (const float*)d_in[9];
  const float* wn_b3  = (const float*)d_in[10];
  const float* wlin   = (const float*)d_in[11];

  // workspace layout (all region sizes are multiples of 256 bytes)
  char* ws = (char*)d_ws;
  size_t off = 0;
  int*    knn      = (int*)(ws + off);    off += (size_t)B_ * S_ * K_ * sizeof(int);  // 2 MB
  __bf16* w1b      = (__bf16*)(ws + off); off += (size_t)W1N * 2;                     // 40 KB
  __bf16* w2b      = (__bf16*)(ws + off); off += (size_t)W2N * 2;                     // 32 KB
  __bf16* w3b      = (__bf16*)(ws + off); off += (size_t)W3N * 2;                     // 64 KB
  __bf16* wlinb    = (__bf16*)(ws + off); off += (size_t)WLN * 2;                     // 1 MB
  __bf16* points_t = (__bf16*)(ws + off); off += (size_t)PTN * 2;                     // 16 MB
  __bf16* flat     = (__bf16*)(ws + off); off += (size_t)B_ * S_ * FLATC_ * 2;        // 64 MB

  float* out_xyz  = (float*)d_out;                      // (B,3,S)
  float* out_feat = out_xyz + (size_t)B_ * 3 * S_;      // (B,256,S)

  pc_newxyz_kernel<<<(B_ * 3 * S_ + 255) / 256, 256, 0, stream>>>(xyz, out_xyz);
  pc_prep_kernel<<<(W1N + W2N + W3N + WLN + 255) / 256, 256, 0, stream>>>(
      w1, w2, w3, wlin, w1b, w2b, w3b, wlinb);
  pc_transpose_kernel<<<(PTN + 255) / 256, 256, 0, stream>>>(points, points_t);
  pc_knn_kernel<<<(B_ * S_) / 256, 256, 0, stream>>>(xyz, knn);
  pc_mlp_kernel<<<B_ * S_, 256, 0, stream>>>(xyz, points_t, knn, w1b, w2b, w3b,
                                             wn_w1, wn_b1, wn_w2, wn_b2, wn_w3, wn_b3, flat);
  pc_final_kernel<<<(B_ * S_) / 16, 256, 0, stream>>>(flat, wlinb, out_feat);
}
